// PAM_Module_11020886082196
// MI455X (gfx1250) — matrix operand order repaired and checked by reading, not compiled, not run on hardware
//
#include <hip/hip_runtime.h>
#include <hip/hip_bf16.h>

typedef __attribute__((ext_vector_type(16))) _Float16 v16h;
typedef __attribute__((ext_vector_type(8)))  float    v8f;
typedef __attribute__((ext_vector_type(4)))  unsigned int u32x4;
typedef __attribute__((ext_vector_type(8)))  int      i32x8;
typedef __attribute__((ext_vector_type(4)))  int      i32x4;

union HF16 {
    v16h v;
    _Float16 h[16];
    uint4 q[2];
};
static_assert(sizeof(HF16) == 32, "frag size");

#define BATCH 4
#define CH    256
#define NPOS  4096
#define CQ    32

// ---------------------------------------------------------------------------
// Kernel 1: concat Wq|Wk|Wv -> f16 (320 x 256), biases -> bcat (320 f32)
// ---------------------------------------------------------------------------
__global__ void pam_wconv(const float* __restrict__ Wq, const float* __restrict__ bq,
                          const float* __restrict__ Wk, const float* __restrict__ bk,
                          const float* __restrict__ Wv, const float* __restrict__ bv,
                          _Float16* __restrict__ Wc16, float* __restrict__ bcat)
{
    int idx = blockIdx.x * 256 + threadIdx.x;
    if (idx >= 320 * 256) return;
    int row = idx >> 8, col = idx & 255;
    float w, bb;
    if (row < 32)       { w = Wq[row * 256 + col];        bb = bq[row]; }
    else if (row < 64)  { w = Wk[(row - 32) * 256 + col]; bb = bk[row - 32]; }
    else                { w = Wv[(row - 64) * 256 + col]; bb = bv[row - 64]; }
    Wc16[idx] = (_Float16)w;
    if (col == 0) bcat[row] = bb;
}

// ---------------------------------------------------------------------------
// Kernel 2: x (B,C,N) f32 -> xT16 (B,N,C) f16 via 32x32 LDS tile
// ---------------------------------------------------------------------------
__global__ void pam_xpose(const float* __restrict__ x, _Float16* __restrict__ xT16)
{
    __shared__ float tile[32][33];
    const int b   = blockIdx.z;
    const int c0  = blockIdx.y * 32;
    const int nn0 = blockIdx.x * 32;
    const int tx = threadIdx.x;   // 0..31
    const int ty = threadIdx.y;   // 0..7
#pragma unroll
    for (int i = 0; i < 4; i++)
        tile[ty + i * 8][tx] = x[((size_t)(b * CH + c0 + ty + i * 8)) * NPOS + nn0 + tx];
    __syncthreads();
#pragma unroll
    for (int i = 0; i < 4; i++)
        xT16[((size_t)(b * NPOS + nn0 + ty + i * 8)) * CH + c0 + tx] =
            (_Float16)tile[tx][ty + i * 8];
}

// ---------------------------------------------------------------------------
// Kernel 3: projection GEMM [320x256] x [256 x N] with WMMA f16 -> q16,k16,v16
// ---------------------------------------------------------------------------
__global__ __launch_bounds__(256, 1) void pam_proj(
    const _Float16* __restrict__ xT16, const _Float16* __restrict__ Wc16,
    const float* __restrict__ bcat,
    _Float16* __restrict__ q16, _Float16* __restrict__ k16, _Float16* __restrict__ v16)
{
    const int b     = blockIdx.y;
    const int wave  = threadIdx.x >> 5;
    const int lane  = threadIdx.x & 31;
    const int lrow  = lane & 15;
    const int lhalf = lane >> 4;
    const int n     = blockIdx.x * 128 + wave * 16 + lrow;

    HF16 bx[8];
#pragma unroll
    for (int kc = 0; kc < 8; kc++) {
        const _Float16* src = xT16 + ((size_t)(b * NPOS + n)) * CH + kc * 32 + lhalf * 8;
        bx[kc].q[0] = *(const uint4*)src;
        bx[kc].q[1] = *(const uint4*)(src + 16);
    }

    for (int rt = 0; rt < 20; rt++) {
        v8f acc = {};
#pragma unroll
        for (int kc = 0; kc < 8; kc++) {
            HF16 aw;
            const _Float16* src = Wc16 + (size_t)(rt * 16 + lrow) * CH + kc * 32 + lhalf * 8;
            aw.q[0] = *(const uint4*)src;
            aw.q[1] = *(const uint4*)(src + 16);
            acc = __builtin_amdgcn_wmma_f32_16x16x32_f16(false, aw.v, false, bx[kc].v,
                                                         (short)0, acc, false, false);
        }
#pragma unroll
        for (int r = 0; r < 8; r++) {
            int row = rt * 16 + r + lhalf * 8;
            _Float16 hv = (_Float16)(acc[r] + bcat[row]);
            if (rt < 2)       q16[((size_t)(b * NPOS + n)) * CQ + row] = hv;
            else if (rt < 4)  k16[((size_t)(b * NPOS + n)) * CQ + (row - 32)] = hv;
            else              v16[((size_t)(b * CH + (row - 64))) * NPOS + n] = hv;
        }
    }
}

// ---------------------------------------------------------------------------
// TDM helper: async DMA of the V tile (256 rows x 32 f16, row stride NPOS*2 B)
// global -> LDS with 16B padding after every 64B row (-> 80B LDS row stride).
// D# per CDNA5 ISA ch.8: group0 {count, lds_addr, global_addr, type=2},
// group1 {data_size=2B, pad_enable, pad_interval=64B, pad_amount=16B,
//         tensor_dims, tile_dims, dim0 stride}, groups 2/3 zero (2D tile).
// Toolchain uses the 6-arg builtin (extra SGPR group, zero-filled).
// ---------------------------------------------------------------------------
__device__ __forceinline__ void tdm_load_vtile(const _Float16* gsrc, unsigned lds_off)
{
    unsigned long long ga = (unsigned long long)(size_t)gsrc;
    u32x4 g0 = { 1u,                                   // count = 1 valid descriptor
                 lds_off,                              // LDS destination (bytes)
                 (unsigned)(ga & 0xFFFFFFFFu),         // global_addr[31:0]
                 (unsigned)((ga >> 32) & 0x01FFFFFFu) | (2u << 30) };  // addr[56:32] | type=2
    i32x8 g1 = { (int)((1u << 16)                      // data_size = 2 bytes
                     | (1u << 20)                      // pad_enable
                     | (3u << 22)                      // pad_interval: 16 DWORDs = 64 B
                     | (3u << 25)),                    // pad_amount:   4 DWORDs = 16 B
                 (int)(4096u << 16),                   // tensor_dim0 = 4096 (low 16 @ bit48)
                 (int)(256u << 16),                    // tensor_dim0 hi | tensor_dim1 = 256
                 (int)(32u << 16),                     // tensor_dim1 hi | tile_dim0 = 32
                 (int)(256u),                          // tile_dim1 = 256, tile_dim2 = 0
                 (int)(4096u),                         // tensor_dim0_stride = 4096 elems
                 0, 0 };
    i32x4 z4 = { 0, 0, 0, 0 };
    i32x8 z8 = { 0, 0, 0, 0, 0, 0, 0, 0 };
    __builtin_amdgcn_tensor_load_to_lds(g0, g1, z4, z4, z8, 0);
}

// ---------------------------------------------------------------------------
// Kernel 4: flash attention, double-buffered. 4 waves/block, 16 queries/wave,
// 32-key blocks. V tile staged by the Tensor Data Mover (async, TENSORcnt),
// k tile prefetched to regs during compute. E^T via WMMA, O^T = V * P^T.
// ---------------------------------------------------------------------------
__global__ __launch_bounds__(128, 1) void pam_attn(
    const _Float16* __restrict__ q16, const _Float16* __restrict__ k16,
    const _Float16* __restrict__ v16, const float* __restrict__ x,
    const float* __restrict__ gamma, float* __restrict__ out)
{
    __shared__ __align__(16) _Float16 lds_k[2][32 * 32];   // [buf][key][chan]
    __shared__ __align__(16) _Float16 lds_v[2][256 * 40];  // [buf][chan][key], 80B rows

    const int b     = blockIdx.y;
    const int wave  = threadIdx.x >> 5;
    const int lane  = threadIdx.x & 31;
    const int lrow  = lane & 15;
    const int lhalf = lane >> 4;
    const int m0    = blockIdx.x * 64 + wave * 16;

    const _Float16* vbase = v16 + ((size_t)b * CH) * NPOS;

    // q^T B-fragment (32 chans x 16 queries), loop invariant
    HF16 bq;
    {
        const _Float16* src = q16 + ((size_t)(b * NPOS + m0 + lrow)) * CQ + lhalf * 8;
        bq.q[0] = *(const uint4*)src;
        bq.q[1] = *(const uint4*)(src + 16);
    }

    float run_m = -3.0e38f, run_l = 0.0f;
    v8f o[16];
#pragma unroll
    for (int i = 0; i < 16; i++) o[i] = (v8f){};

    const int kkey  = threadIdx.x >> 2;         // k-stage: 128 threads x 16B
    const int kpart = threadIdx.x & 3;

    // ---- prologue: stage block 0 into buffer 0 ----
    if (wave == 0)
        tdm_load_vtile(vbase, (unsigned)(size_t)(void*)&lds_v[0][0]);
    {
        uint4 kr = *((const uint4*)(k16 + ((size_t)(b * NPOS + kkey)) * CQ) + kpart);
        *((uint4*)(lds_k[0] + kkey * 32) + kpart) = kr;
    }
    if (wave == 0) __builtin_amdgcn_s_wait_tensorcnt(0);
    __syncthreads();

    int cur = 0;
    for (int i = 0; i < NPOS / 32; i++) {
        const int n0 = i * 32;
        const int nxt = cur ^ 1;
        const bool havenext = (i + 1) < (NPOS / 32);

        // prefetch next block: TDM for V, registers for k (overlaps with WMMAs)
        uint4 kreg = {};
        if (havenext) {
            if (wave == 0)
                tdm_load_vtile(vbase + (n0 + 32),
                               (unsigned)(size_t)(void*)&lds_v[nxt][0]);
            kreg = *((const uint4*)(k16 + ((size_t)(b * NPOS + n0 + 32 + kkey)) * CQ) + kpart);
        }

        // E^T tiles: A = key rows (16x32) from LDS, B = q^T -> D[key][query]
        HF16 a0, a1;
        {
            const _Float16* base = lds_k[cur] + lrow * 32 + lhalf * 8;
            a0.q[0] = *(const uint4*)base;
            a0.q[1] = *(const uint4*)(base + 16);
            a1.q[0] = *(const uint4*)(base + 16 * 32);
            a1.q[1] = *(const uint4*)(base + 16 * 32 + 16);
        }
        v8f zc = {};
        v8f et0 = __builtin_amdgcn_wmma_f32_16x16x32_f16(false, a0.v, false, bq.v,
                                                         (short)0, zc, false, false);
        v8f et1 = __builtin_amdgcn_wmma_f32_16x16x32_f16(false, a1.v, false, bq.v,
                                                         (short)0, zc, false, false);

        // online softmax over keys for this lane's query column
        float tmax = et0[0];
#pragma unroll
        for (int r = 1; r < 8; r++) tmax = fmaxf(tmax, et0[r]);
#pragma unroll
        for (int r = 0; r < 8; r++) tmax = fmaxf(tmax, et1[r]);
        tmax = fmaxf(tmax, __shfl_xor(tmax, 16, 32));
        float newm  = fmaxf(run_m, tmax);
        float alpha = __expf(run_m - newm);
        float p0[8], p1[8], psum = 0.0f;
#pragma unroll
        for (int r = 0; r < 8; r++) {
            p0[r] = __expf(et0[r] - newm);
            p1[r] = __expf(et1[r] - newm);
            psum += p0[r] + p1[r];
        }
        psum += __shfl_xor(psum, 16, 32);
        run_l = run_l * alpha + psum;
        run_m = newm;
#pragma unroll
        for (int j = 0; j < 16; j++)
#pragma unroll
            for (int r = 0; r < 8; r++) o[j][r] *= alpha;

        // P^T (D layout) -> B-fragment (32 keys x 16 queries) f16
        HF16 pb;
#pragma unroll
        for (int r = 0; r < 8; r++) {
            float s0 = __shfl_xor(p0[r], 16, 32);
            float s1 = __shfl_xor(p1[r], 16, 32);
            pb.h[r]     = (_Float16)((lhalf == 0) ? p0[r] : s1);
            pb.h[8 + r] = (_Float16)((lhalf == 0) ? s0    : p1[r]);
        }

        // O^T += V(16c x 32k) x P^T(32k x 16q), 16 channel groups
#pragma unroll
        for (int cg = 0; cg < 16; cg++) {
            HF16 va;
            const _Float16* vb = lds_v[cur] + (cg * 16 + lrow) * 40 + lhalf * 16;
            va.q[0] = *(const uint4*)vb;
            va.q[1] = *(const uint4*)(vb + 8);
            o[cg] = __builtin_amdgcn_wmma_f32_16x16x32_f16(false, va.v, false, pb.v,
                                                           (short)0, o[cg], false, false);
        }

        // publish next buffer: k regs -> LDS, TDM drained, then barrier
        if (havenext) {
            *((uint4*)(lds_k[nxt] + kkey * 32) + kpart) = kreg;
            if (wave == 0) __builtin_amdgcn_s_wait_tensorcnt(0);
        }
        __syncthreads();
        cur = nxt;
    }

    // epilogue: out = gamma * (O / l) + x
    float inv_l = 1.0f / run_l;
    float g = gamma[0];
#pragma unroll
    for (int cg = 0; cg < 16; cg++) {
#pragma unroll
        for (int r = 0; r < 8; r++) {
            int c = cg * 16 + r + lhalf * 8;
            size_t idx = ((size_t)(b * CH + c)) * NPOS + (m0 + lrow);
            out[idx] = g * (o[cg][r] * inv_l) + x[idx];
        }
    }
}

// ---------------------------------------------------------------------------
extern "C" void kernel_launch(void* const* d_in, const int* in_sizes, int n_in,
                              void* d_out, int out_size, void* d_ws, size_t ws_size,
                              hipStream_t stream)
{
    (void)in_sizes; (void)n_in; (void)out_size; (void)ws_size;
    const float* x     = (const float*)d_in[0];
    const float* Wq    = (const float*)d_in[1];
    const float* bq    = (const float*)d_in[2];
    const float* Wk    = (const float*)d_in[3];
    const float* bk    = (const float*)d_in[4];
    const float* Wv    = (const float*)d_in[5];
    const float* bv    = (const float*)d_in[6];
    const float* gamma = (const float*)d_in[7];
    float* out = (float*)d_out;

    char* ws = (char*)d_ws;
    _Float16* xT16 = (_Float16*)(ws);                          //  8 MB
    _Float16* Wc16 = (_Float16*)(ws + (9u  << 20));            //  160 KB
    float*    bcat = (float*)   (ws + (9u  << 20) + (512u << 10));
    _Float16* q16  = (_Float16*)(ws + (10u << 20));            //  1 MB
    _Float16* k16  = (_Float16*)(ws + (11u << 20));            //  1 MB
    _Float16* v16  = (_Float16*)(ws + (12u << 20));            //  8 MB

    pam_wconv<<<dim3(320), dim3(256), 0, stream>>>(Wq, bq, Wk, bk, Wv, bv, Wc16, bcat);
    pam_xpose<<<dim3(128, 8, 4), dim3(32, 8), 0, stream>>>(x, xT16);
    pam_proj <<<dim3(32, 4),  dim3(256), 0, stream>>>(xT16, Wc16, bcat, q16, k16, v16);
    pam_attn <<<dim3(64, 4),  dim3(128), 0, stream>>>(q16, k16, v16, x, gamma, out);
}
